// graph_constructor_timestamp_22728966930661
// MI455X (gfx1250) — compile-verified, matrix operand order checked by reading, not run on hardware
//
#include <hip/hip_runtime.h>
#include <hip/hip_fp16.h>

typedef _Float16 half_t;
typedef __attribute__((ext_vector_type(8)))  _Float16 v8h;
typedef __attribute__((ext_vector_type(16))) _Float16 v16h;
typedef __attribute__((ext_vector_type(8)))  float    v8f;
typedef __attribute__((ext_vector_type(4)))  float    v4f;

#define ALPHA 3.0f
#define NROW  2048
#define DDIM  64
#define TOPK  32
#define TSTEP 32

// ---------------------------------------------------------------------------
// Kernel A: nv[n][d] = tanh(ALPHA * (emb[idx[n]] . w[d,:] + b[d])), stored f16
// 2048x64 output, trivial FLOPs; w (16KB) is L2 resident.
// ---------------------------------------------------------------------------
__global__ void nv_kernel(const int* __restrict__ idx,
                          const float* __restrict__ emb,
                          const float* __restrict__ w,
                          const float* __restrict__ b,
                          half_t* __restrict__ nvh) {
    int t = blockIdx.x * blockDim.x + threadIdx.x;   // [0, 2048*64)
    int d = t & (DDIM - 1);
    int n = t >> 6;
    int row = idx[n];
    const float* e  = emb + (long)row * DDIM;
    const float* wr = w + (long)d * DDIM;            // out = emb @ w.T
    float acc = b[d];
#pragma unroll
    for (int k = 0; k < DDIM; k += 4) {
        v4f ev = *(const v4f*)(e + k);
        v4f wv = *(const v4f*)(wr + k);
        acc = fmaf(ev.x, wv.x, acc);
        acc = fmaf(ev.y, wv.y, acc);
        acc = fmaf(ev.z, wv.z, acc);
        acc = fmaf(ev.w, wv.w, acc);
    }
    nvh[t] = (half_t)tanhf(ALPHA * acc);
}

// ---------------------------------------------------------------------------
// Kernel B: adj = relu(tanh(ALPHA * (nv1@nv2^T - nv2@nv1^T))) via WMMA.
// One wave32 per 16x16 tile; K=64 done as two v_wmma_f32_16x16x32_f16 steps
// per accumulator. A-fragment: lane L holds row (tileI + L%16), halves
// h=0..7 -> k = k0 + (L/16)*8 + h ; h=8..15 -> k = k0+16 + (L/16)*8 + (h-8).
// B-fragment is the same gather applied to the J-tile rows (B = nv^T).
// ---------------------------------------------------------------------------
__device__ __forceinline__ v16h load_frag(const half_t* __restrict__ base,
                                          int row, int k0, int hk) {
    const half_t* p = base + (long)row * DDIM + k0 + hk;
    v8h lo = *(const v8h*)(p);        // k0+hk   .. +7   (16B aligned)
    v8h hi = *(const v8h*)(p + 16);   // k0+16+hk .. +7
    return __builtin_shufflevector(lo, hi, 0, 1, 2, 3, 4, 5, 6, 7,
                                            8, 9, 10, 11, 12, 13, 14, 15);
}

__global__ void adj_wmma_kernel(const half_t* __restrict__ nv1h,
                                const half_t* __restrict__ nv2h,
                                float* __restrict__ adj) {
    int wave = (blockIdx.x * blockDim.x + threadIdx.x) >> 5;  // [0, 128*128)
    int lane = threadIdx.x & 31;
    int ti = wave >> 7;
    int tj = wave & 127;
    int I = ti << 4, J = tj << 4;
    int r  = lane & 15;          // row-in-tile for both A and B gathers
    int hk = (lane >> 4) * 8;    // upper half-wave takes K+8

    v8f acc1 = {};   // nv1 @ nv2^T
    v8f acc2 = {};   // nv2 @ nv1^T
#pragma unroll
    for (int k0 = 0; k0 < DDIM; k0 += 32) {
        v16h A1 = load_frag(nv1h, I + r, k0, hk);
        v16h A2 = load_frag(nv2h, I + r, k0, hk);
        v16h B1 = load_frag(nv1h, J + r, k0, hk);
        v16h B2 = load_frag(nv2h, J + r, k0, hk);
        acc1 = __builtin_amdgcn_wmma_f32_16x16x32_f16(false, A1, false, B2,
                                                      (short)0, acc1, false, false);
        acc2 = __builtin_amdgcn_wmma_f32_16x16x32_f16(false, A2, false, B1,
                                                      (short)0, acc2, false, false);
    }
    // D layout: VGPR v -> M = v + (lane/16)*8, N = lane%16
    int mbase = (lane >> 4) * 8;
    int n = lane & 15;
#pragma unroll
    for (int v = 0; v < 8; ++v) {
        float a = acc1[v] - acc2[v];
        float o = fmaxf(tanhf(ALPHA * a), 0.0f);
        adj[(long)(I + mbase + v) * NROW + (J + n)] = o;
    }
}

// ---------------------------------------------------------------------------
// Kernel C: per-row top-32 of (adj + 0.01*rand_noise). One block per row;
// scores in LDS, 32 argmax sweeps, result written as a 2048-bit row bitmask.
// ---------------------------------------------------------------------------
__global__ void topk_kernel(const float* __restrict__ adj,
                            const float* __restrict__ rnd,
                            unsigned int* __restrict__ maskbits) {
    __shared__ float s[NROW];
    __shared__ float rv[256];
    __shared__ int   ri[256];
    int row = blockIdx.x;
    int tid = threadIdx.x;
    for (int j = tid; j < NROW; j += 256)
        s[j] = adj[(long)row * NROW + j] + 0.01f * rnd[(long)row * NROW + j];
    if (tid < NROW / 32) maskbits[row * (NROW / 32) + tid] = 0u;
    __syncthreads();

    for (int k = 0; k < TOPK; ++k) {
        float best = -1e30f; int bi = NROW;
        for (int j = tid; j < NROW; j += 256) {
            float v = s[j];
            if (v > best || (v == best && j < bi)) { best = v; bi = j; }
        }
        rv[tid] = best; ri[tid] = bi;
        __syncthreads();
        for (int off = 128; off > 0; off >>= 1) {
            if (tid < off) {
                float ov = rv[tid + off]; int oi = ri[tid + off];
                if (ov > rv[tid] || (ov == rv[tid] && oi < ri[tid])) {
                    rv[tid] = ov; ri[tid] = oi;
                }
            }
            __syncthreads();
        }
        if (tid == 0) {
            int w = ri[0];
            s[w] = -1e30f;
            maskbits[row * (NROW / 32) + (w >> 5)] |= (1u << (w & 31));
        }
        __syncthreads();
    }
}

// ---------------------------------------------------------------------------
// Kernel E: out[t,i,j] = mask ? relu(tanh(ALPHA*(adj[i,j] + 0.01*noise))) : 0.
// Float4 per thread. Mask word (512KB, L2-hot) checked BEFORE loading noise:
// ~94% of quads skip the 537MB noise stream entirely. Non-temporal hints on
// the streaming noise/out traffic keep adj+mask resident in L2.
// ---------------------------------------------------------------------------
__global__ void out_kernel(const float* __restrict__ adj,
                           const unsigned int* __restrict__ maskbits,
                           const float* __restrict__ noise,
                           float* __restrict__ out) {
    int g = blockIdx.x * blockDim.x + threadIdx.x;   // quad index < 2^25
    long e = (long)g << 2;                           // element index
    int j = (int)(e & (NROW - 1));                   // multiple of 4
    int i = (int)((e >> 11) & (NROW - 1));

    unsigned mw = maskbits[i * (NROW / 32) + (j >> 5)];
    unsigned m4 = (mw >> (j & 31)) & 0xFu;           // 4 bits, never split

    v4f o = {0.0f, 0.0f, 0.0f, 0.0f};
    if (m4) {
        v4f nz = __builtin_nontemporal_load((const v4f*)(noise + e));
        v4f av = *(const v4f*)(adj + (long)i * NROW + j);
        if (m4 & 1u) o.x = fmaxf(tanhf(ALPHA * fmaf(0.01f, nz.x, av.x)), 0.0f);
        if (m4 & 2u) o.y = fmaxf(tanhf(ALPHA * fmaf(0.01f, nz.y, av.y)), 0.0f);
        if (m4 & 4u) o.z = fmaxf(tanhf(ALPHA * fmaf(0.01f, nz.z, av.z)), 0.0f);
        if (m4 & 8u) o.w = fmaxf(tanhf(ALPHA * fmaf(0.01f, nz.w, av.w)), 0.0f);
    }
    __builtin_nontemporal_store(o, (v4f*)(out + e));
}

// ---------------------------------------------------------------------------
extern "C" void kernel_launch(void* const* d_in, const int* in_sizes, int n_in,
                              void* d_out, int out_size, void* d_ws, size_t ws_size,
                              hipStream_t stream) {
    const int*   idx  = (const int*)d_in[0];
    const float* emb1 = (const float*)d_in[1];
    const float* emb2 = (const float*)d_in[2];
    const float* w1   = (const float*)d_in[3];
    const float* b1   = (const float*)d_in[4];
    const float* w2   = (const float*)d_in[5];
    const float* b2   = (const float*)d_in[6];
    const float* rnd  = (const float*)d_in[7];   // (N, N)
    const float* rnn  = (const float*)d_in[8];   // (T, N, N)
    float* out = (float*)d_out;

    char* ws = (char*)d_ws;
    half_t*       nv1h = (half_t*)ws;                              // 256 KB
    half_t*       nv2h = (half_t*)(ws + (256 << 10));              // 256 KB
    float*        adj  = (float*)(ws + (512 << 10));               // 16 MB
    unsigned int* mbit = (unsigned int*)(ws + (512 << 10) + (16 << 20)); // 512 KB

    // A: nv1, nv2 (2048x64 each)
    nv_kernel<<<(NROW * DDIM) / 256, 256, 0, stream>>>(idx, emb1, w1, b1, nv1h);
    nv_kernel<<<(NROW * DDIM) / 256, 256, 0, stream>>>(idx, emb2, w2, b2, nv2h);

    // B: 128x128 tiles of 16x16, 8 waves per 256-thread block
    adj_wmma_kernel<<<(128 * 128) / 8, 256, 0, stream>>>(nv1h, nv2h, adj);

    // C: top-32 per row -> bitmask
    topk_kernel<<<NROW, 256, 0, stream>>>(adj, rnd, mbit);

    // E: masked streaming epilogue over (T, N, N), float4 per thread
    long quads = (long)TSTEP * NROW * NROW / 4;     // 33,554,432
    out_kernel<<<(int)(quads / 256), 256, 0, stream>>>(adj, mbit, rnn, out);
}
